// GATJK_61443802137320
// MI455X (gfx1250) — matrix operand order repaired and checked by reading, not compiled, not run on hardware
//
#include <hip/hip_runtime.h>
#include <hip/hip_bf16.h>

// ---------------------------------------------------------------------------
// Types for WMMA / TDM
// ---------------------------------------------------------------------------
typedef __attribute__((ext_vector_type(16))) __bf16 v16bf;
typedef __attribute__((ext_vector_type(8)))  __bf16 v8bf;
typedef __attribute__((ext_vector_type(8)))  float  v8f;
typedef __attribute__((ext_vector_type(4)))  unsigned u32x4;
typedef __attribute__((ext_vector_type(8)))  int      i32x8;
typedef __attribute__((ext_vector_type(4)))  int      i32x4;

#define HC 256   // H*C
#define NH 4     // heads
#define CC 64    // channels per head

// LDS A tile: 64 rows x 32 bf16, padded by TDM to 40-element (80B) row stride
#define A_ROWS   64
#define A_KSTEP  32
#define A_STRIDE 40          // elements (32 data + 8 pad), conflict-free reads

// ---------------------------------------------------------------------------
// util: fill u32 (zeroing f32 buffers and encoded-max buffers)
// ---------------------------------------------------------------------------
__global__ void gatjk_fill_u32(unsigned* __restrict__ p, unsigned v, int n) {
    int i = blockIdx.x * blockDim.x + threadIdx.x;
    int stride = gridDim.x * blockDim.x;
    for (; i < n; i += stride) p[i] = v;
}

// ---------------------------------------------------------------------------
// f32 -> bf16 convert
// ---------------------------------------------------------------------------
__global__ void gatjk_f32_to_bf16(const float* __restrict__ s, __bf16* __restrict__ d, int n) {
    int i = blockIdx.x * blockDim.x + threadIdx.x;
    int stride = gridDim.x * blockDim.x;
    for (; i < n; i += stride) d[i] = (__bf16)s[i];
}

// ---------------------------------------------------------------------------
// Pack weight [K,N] f32 row-major into per-lane WMMA B-fragment order (bf16).
// For tile (kt,nt): lane L holds 16 contiguous bf16: n = nt*16+(L&15),
// k = kt*32 + ((L&16)?16:0) + e   (contiguous k order, same as the A reads)
// ---------------------------------------------------------------------------
__global__ void gatjk_pack_w(const float* __restrict__ W, __bf16* __restrict__ P,
                             int K, int Nc) {
    int idx = blockIdx.x * blockDim.x + threadIdx.x;
    int total = K * Nc;
    if (idx >= total) return;
    int e     = idx & 15;
    int lane  = (idx >> 4) & 31;
    int rest  = idx >> 9;                 // kt*ntiles + nt
    int ntiles = Nc >> 4;
    int nt = rest % ntiles;
    int kt = rest / ntiles;
    int k = kt * 32 + ((lane & 16) ? 16 : 0) + e;
    int n = nt * 16 + (lane & 15);
    P[idx] = (__bf16)W[k * Nc + n];
}

// ---------------------------------------------------------------------------
// TDM: DMA a 64x32 bf16 tile (row-major, row stride = strideK elements) from
// global into LDS at lds_addr, with 16B padding appended per 64B row
// (pad_interval=3 -> every 16 DWORDs, pad_amount=3 -> 4 DWORDs).
// Rows/cols beyond (dim1_rem, dim0_rem) are zero-filled by hardware.
// 6-arg builtin (clang-23 / therock signature); descriptor per CDNA5 ISA
// ch. 10.8 (D# groups 0..3).
// ---------------------------------------------------------------------------
__device__ __forceinline__ void tdm_load_a_tile(unsigned lds_addr,
                                                const __bf16* gptr,
                                                int dim0_rem, int dim1_rem,
                                                int strideK) {
    unsigned long long ga = (unsigned long long)(size_t)gptr;
    u32x4 g0;
    g0[0] = 1u;                                              // count=1, load
    g0[1] = lds_addr;                                        // LDS byte addr
    g0[2] = (unsigned)(ga & 0xFFFFFFFFull);                  // global addr lo
    g0[3] = (unsigned)((ga >> 32) & 0x01FFFFFFull) | (2u << 30); // hi | type=2
    i32x8 g1;
    g1[0] = (int)((1u << 16)            // data_size = 2 bytes
                | (1u << 20)            // pad_enable
                | (3u << 22)            // pad_interval: 16 DWORDs (64B row)
                | (3u << 25));          // pad_amount: 4 DWORDs (16B)
    g1[1] = (int)(((unsigned)dim0_rem & 0xFFFFu) << 16);     // tensor_dim0 lo16
    g1[2] = (int)(((unsigned)dim0_rem >> 16)
                | (((unsigned)dim1_rem & 0xFFFFu) << 16));   // dim0 hi | dim1 lo
    g1[3] = (int)(((unsigned)dim1_rem >> 16)
                | ((unsigned)A_KSTEP << 16));                // dim1 hi | tile_dim0
    g1[4] = A_ROWS;                                          // tile_dim1 (tile_dim2=0)
    g1[5] = strideK;                                         // tensor_dim0_stride
    g1[6] = 0;
    g1[7] = 0;
    i32x4 z4 = {0, 0, 0, 0};
    i32x8 z8 = {0, 0, 0, 0, 0, 0, 0, 0};
    __builtin_amdgcn_tensor_load_to_lds(g0, g1, z4, z4, z8, 0);
}

// ---------------------------------------------------------------------------
// GEMM: C[M,Nc] = A[M,K](bf16,row-major) * B(packed bf16) [+ bias]
// block = 128 threads = 4 waves; block tile 64(M) x 64(N).
// Wave w owns n-tile w and all 4 m-tiles -> 4 accumulators, 4 WMMA / K-step.
// A tiles DMA'd by the Tensor Data Mover into double-buffered LDS.
// ---------------------------------------------------------------------------
__global__ __launch_bounds__(128) void gatjk_gemm_wmma(
    const __bf16* __restrict__ A, int M, int K,
    const __bf16* __restrict__ Bp,
    float* __restrict__ C, int Nc,
    const float* __restrict__ bias) {

    __shared__ __align__(16) __bf16 As[2][A_ROWS * A_STRIDE];

    const int m0     = blockIdx.x * A_ROWS;
    const int wave   = threadIdx.x >> 5;
    const int lane   = threadIdx.x & 31;
    const int ntile  = blockIdx.y * 4 + wave;
    const int n0     = ntile * 16;
    const int ntiles = Nc >> 4;
    const int kSteps = K >> 5;
    const int dim1_rem = M - m0;             // TDM zero-fills rows beyond this

    v8f acc0 = {}, acc1 = {}, acc2 = {}, acc3 = {};

    // prologue: DMA first A tile
    if (wave == 0) {
        tdm_load_a_tile((unsigned)(size_t)&As[0][0],
                        A + (size_t)m0 * K, K, dim1_rem, K);
    }

    for (int kt = 0; kt < kSteps; ++kt) {
        __syncthreads();   // all waves done reading the buffer we overwrite next
        if (wave == 0) {
            if (kt + 1 < kSteps) {
                tdm_load_a_tile((unsigned)(size_t)&As[(kt + 1) & 1][0],
                                A + (size_t)m0 * K + (kt + 1) * A_KSTEP,
                                K - (kt + 1) * A_KSTEP, dim1_rem, K);
                __builtin_amdgcn_s_wait_tensorcnt(1);  // current tile landed
            } else {
                __builtin_amdgcn_s_wait_tensorcnt(0);
            }
        }
        __syncthreads();   // buf[kt&1] visible to all waves

        // B fragment: 32 contiguous bytes per lane from packed weights
        const __bf16* bp = Bp + ((size_t)(kt * ntiles + ntile) * 32 + lane) * 16;
        v8bf blo = *(const v8bf*)(bp);
        v8bf bhi = *(const v8bf*)(bp + 8);
        v16bf b;
#pragma unroll
        for (int i = 0; i < 8; ++i) { b[i] = blo[i]; b[i + 8] = bhi[i]; }

        const __bf16* abase = &As[kt & 1][0];
        const int am  = lane & 15;
        const int abk = (lane >> 4) << 4;            // 0 or 16 (K-half select)
#pragma unroll
        for (int mt = 0; mt < 4; ++mt) {
            const __bf16* ar = abase + (mt * 16 + am) * A_STRIDE + abk;
            v8bf alo = *(const v8bf*)(ar);           // K = abk..abk+7
            v8bf ahi = *(const v8bf*)(ar + 8);       // K = abk+8..abk+15
            v16bf a;
#pragma unroll
            for (int i = 0; i < 8; ++i) { a[i] = alo[i]; a[i + 8] = ahi[i]; }
            v8f& acc = (mt == 0) ? acc0 : (mt == 1) ? acc1 : (mt == 2) ? acc2 : acc3;
            acc = __builtin_amdgcn_wmma_f32_16x16x32_bf16(
                false, a, false, b, (short)0, acc, false, false);
        }
    }

    // C/D layout: VGPR r -> M = r + 8*(lane>=16); N = lane&15
    const int n  = n0 + (lane & 15);
    const float bv = bias ? bias[n] : 0.0f;
#pragma unroll
    for (int mt = 0; mt < 4; ++mt) {
        v8f& acc = (mt == 0) ? acc0 : (mt == 1) ? acc1 : (mt == 2) ? acc2 : acc3;
        int mbase = m0 + mt * 16 + ((lane >> 4) << 3);
#pragma unroll
        for (int r = 0; r < 8; ++r) {
            int row = mbase + r;
            if (row < M) C[(size_t)row * Nc + n] = acc[r] + bv;
        }
    }
}

// ---------------------------------------------------------------------------
// alpha_src/alpha_dst: one wave per (node, head); 2 channels per lane,
// wave32 shuffle reduction.
// ---------------------------------------------------------------------------
__global__ void gatjk_alpha(const float* __restrict__ h,
                            const float* __restrict__ a_src,
                            const float* __restrict__ a_dst,
                            float* __restrict__ as, float* __restrict__ ad, int Nn) {
    int wid  = (blockIdx.x * blockDim.x + threadIdx.x) >> 5;
    int lane = threadIdx.x & 31;
    int node = wid >> 2;
    int head = wid & 3;
    if (node >= Nn) return;
    const float* hp  = h + (size_t)node * HC + head * CC;
    const float* avS = a_src + head * CC;
    const float* avD = a_dst + head * CC;
    float h0 = hp[lane], h1 = hp[lane + 32];
    float vs = h0 * avS[lane] + h1 * avS[lane + 32];
    float vd = h0 * avD[lane] + h1 * avD[lane + 32];
#pragma unroll
    for (int off = 16; off; off >>= 1) {
        vs += __shfl_xor(vs, off, 32);
        vd += __shfl_xor(vd, off, 32);
    }
    if (lane == 0) { as[node * NH + head] = vs; ad[node * NH + head] = vd; }
}

// ---------------------------------------------------------------------------
// monotonic float<->uint encoding for atomic segment-max
// ---------------------------------------------------------------------------
__device__ __forceinline__ unsigned enc_f32(float f) {
    unsigned u = __float_as_uint(f);
    return (u & 0x80000000u) ? ~u : (u | 0x80000000u);
}
__device__ __forceinline__ float dec_f32(unsigned e) {
    unsigned u = (e & 0x80000000u) ? (e ^ 0x80000000u) : ~e;
    return __uint_as_float(u);
}
__device__ __forceinline__ float edge_logit(const float* as, const float* ad,
                                            int s, int d, int hh) {
    float v = as[s * NH + hh] + ad[d * NH + hh];
    return v > 0.0f ? v : 0.2f * v;      // leaky_relu, slope 0.2
}
__device__ __forceinline__ void edge_sd(const int* ei, int E, int e, int& s, int& d) {
    if (e < E) { s = ei[e]; d = ei[E + e]; } else { s = e - E; d = e - E; }
}

// pass 1: segment max over incoming edges (encoded atomicMax)
__global__ void gatjk_edge_max(const int* __restrict__ ei, int E, int nE,
                               const float* __restrict__ as, const float* __restrict__ ad,
                               unsigned* __restrict__ menc) {
    int idx = blockIdx.x * blockDim.x + threadIdx.x;
    if (idx >= nE * NH) return;
    int e = idx >> 2, hh = idx & 3, s, d;
    edge_sd(ei, E, e, s, d);
    atomicMax(&menc[d * NH + hh], enc_f32(edge_logit(as, ad, s, d, hh)));
}

// pass 2: e_exp = exp(v - max); segment sum
__global__ void gatjk_edge_exp(const int* __restrict__ ei, int E, int nE,
                               const float* __restrict__ as, const float* __restrict__ ad,
                               const unsigned* __restrict__ menc,
                               float* __restrict__ eexp, float* __restrict__ ssum) {
    int idx = blockIdx.x * blockDim.x + threadIdx.x;
    if (idx >= nE * NH) return;
    int e = idx >> 2, hh = idx & 3, s, d;
    edge_sd(ei, E, e, s, d);
    float v  = edge_logit(as, ad, s, d, hh);
    float ex = __expf(v - dec_f32(menc[d * NH + hh]));
    eexp[idx] = ex;
    atomicAdd(&ssum[d * NH + hh], ex);
}

// pass 3: agg[dst] += h[src] * alpha   (one block per edge, 256 lanes = H*C)
__global__ __launch_bounds__(256) void gatjk_edge_aggregate(
    const int* __restrict__ ei, int E, int nE,
    const float* __restrict__ h,
    const float* __restrict__ eexp, const float* __restrict__ ssum,
    float* __restrict__ agg) {
    int e = blockIdx.x;
    if (e >= nE) return;
    int t = threadIdx.x, s, d;
    edge_sd(ei, E, e, s, d);
    int hh = t >> 6;
    float alpha = eexp[e * NH + hh] / (ssum[d * NH + hh] + 1e-16f);
    atomicAdd(&agg[(size_t)d * HC + t], h[(size_t)s * HC + t] * alpha);
}

// ---------------------------------------------------------------------------
// x1 = elu( (agg + b0) * gamma/sqrt(1+eps) + beta )
// ---------------------------------------------------------------------------
__global__ void gatjk_bias_bn_elu(const float* __restrict__ agg,
                                  const float* __restrict__ b,
                                  const float* __restrict__ gamma,
                                  const float* __restrict__ beta,
                                  float* __restrict__ x1, int n) {
    int i = blockIdx.x * blockDim.x + threadIdx.x;
    int stride = gridDim.x * blockDim.x;
    const float inv = rsqrtf(1.0f + 1e-5f);
    for (; i < n; i += stride) {
        int c = i & (HC - 1);
        float v = (agg[i] + b[c]) * (gamma[c] * inv) + beta[c];
        x1[i] = v > 0.0f ? v : (__expf(v) - 1.0f);
    }
}

// ---------------------------------------------------------------------------
// xjk = max(x1, agg + b1) -> bf16 (feeds final WMMA GEMM)
// ---------------------------------------------------------------------------
__global__ void gatjk_jk_max_bf16(const float* __restrict__ agg,
                                  const float* __restrict__ b1,
                                  const float* __restrict__ x1,
                                  __bf16* __restrict__ out, int n) {
    int i = blockIdx.x * blockDim.x + threadIdx.x;
    int stride = gridDim.x * blockDim.x;
    for (; i < n; i += stride) {
        int c = i & (HC - 1);
        out[i] = (__bf16)fmaxf(x1[i], agg[i] + b1[c]);
    }
}

// ---------------------------------------------------------------------------
// host launcher
// ---------------------------------------------------------------------------
extern "C" void kernel_launch(void* const* d_in, const int* in_sizes, int n_in,
                              void* d_out, int out_size, void* d_ws, size_t ws_size,
                              hipStream_t stream) {
    const float* x      = (const float*)d_in[0];
    const int*   ei     = (const int*)  d_in[1];
    const float* W0     = (const float*)d_in[2];
    const float* a_src0 = (const float*)d_in[3];
    const float* a_dst0 = (const float*)d_in[4];
    const float* b0     = (const float*)d_in[5];
    const float* gamma0 = (const float*)d_in[6];
    const float* beta0  = (const float*)d_in[7];
    const float* W1     = (const float*)d_in[8];
    const float* a_src1 = (const float*)d_in[9];
    const float* a_dst1 = (const float*)d_in[10];
    const float* b1     = (const float*)d_in[11];
    const float* Wf     = (const float*)d_in[12];
    const float* bf     = (const float*)d_in[13];

    const int Nn = in_sizes[0] / 128;     // 50000
    const int E  = in_sizes[1] / 2;       // 800000
    const int nE = E + Nn;                // with self loops

    // workspace carving
    char* ws = (char*)d_ws;
    size_t off = 0;
    auto carve = [&](size_t bytes) -> void* {
        void* p = ws + off;
        off += (bytes + 255) & ~(size_t)255;
        return p;
    };
    float*    h     = (float*)   carve((size_t)Nn * HC * 4);
    float*    agg   = (float*)   carve((size_t)Nn * HC * 4);
    float*    x1    = (float*)   carve((size_t)Nn * HC * 4);
    __bf16*   bfA   = (__bf16*)  carve((size_t)Nn * HC * 2);
    __bf16*   packB = (__bf16*)  carve((size_t)HC * HC * 2);
    float*    asrc  = (float*)   carve((size_t)Nn * NH * 4);
    float*    adst  = (float*)   carve((size_t)Nn * NH * 4);
    unsigned* menc  = (unsigned*)carve((size_t)Nn * NH * 4);
    float*    ssum  = (float*)   carve((size_t)Nn * NH * 4);
    float*    eexp  = (float*)   carve((size_t)nE * NH * 4);
    (void)ws_size; (void)n_in; (void)out_size;

    const int TB = 256;
    const int nNodeHead = Nn * NH;                       // 200000
    const int nEdgeHead = nE * NH;                       // 3.4M
    const int nFeat     = Nn * HC;                       // 12.8M
    const int mBlocks   = (Nn + A_ROWS - 1) / A_ROWS;    // 782
    dim3 gemmGrid256(mBlocks, HC / 64);                  // 782 x 4
    dim3 gemmGridOut(mBlocks, 1);                        // final: N=64

    auto run_edge_phase = [&](const float* aS, const float* aD) {
        gatjk_fill_u32<<<512, TB, 0, stream>>>((unsigned*)menc, 0u, nNodeHead);
        gatjk_fill_u32<<<512, TB, 0, stream>>>((unsigned*)ssum, 0u, nNodeHead);
        gatjk_alpha<<<(nNodeHead * 32 + TB - 1) / TB, TB, 0, stream>>>(
            h, aS, aD, asrc, adst, Nn);
        gatjk_edge_max<<<(nEdgeHead + TB - 1) / TB, TB, 0, stream>>>(
            ei, E, nE, asrc, adst, menc);
        gatjk_edge_exp<<<(nEdgeHead + TB - 1) / TB, TB, 0, stream>>>(
            ei, E, nE, asrc, adst, menc, eexp, ssum);
        gatjk_fill_u32<<<2048, TB, 0, stream>>>((unsigned*)agg, 0u, nFeat);
        gatjk_edge_aggregate<<<nE, 256, 0, stream>>>(ei, E, nE, h, eexp, ssum, agg);
    };

    // ---------- layer 0 ----------
    gatjk_f32_to_bf16<<<2048, TB, 0, stream>>>(x, bfA, Nn * 128);
    gatjk_pack_w<<<(128 * HC + TB - 1) / TB, TB, 0, stream>>>(W0, packB, 128, HC);
    gatjk_gemm_wmma<<<gemmGrid256, 128, 0, stream>>>(bfA, Nn, 128, packB, h, HC, nullptr);
    run_edge_phase(a_src0, a_dst0);
    gatjk_bias_bn_elu<<<2048, TB, 0, stream>>>(agg, b0, gamma0, beta0, x1, nFeat);

    // ---------- layer 1 ----------
    gatjk_f32_to_bf16<<<2048, TB, 0, stream>>>(x1, bfA, nFeat);
    gatjk_pack_w<<<(HC * HC + TB - 1) / TB, TB, 0, stream>>>(W1, packB, HC, HC);
    gatjk_gemm_wmma<<<gemmGrid256, 128, 0, stream>>>(bfA, Nn, HC, packB, h, HC, nullptr);
    run_edge_phase(a_src1, a_dst1);

    // ---------- JK max + final projection (bias fused into GEMM store) ----------
    gatjk_jk_max_bf16<<<2048, TB, 0, stream>>>(agg, b1, x1, bfA, nFeat);
    gatjk_pack_w<<<(HC * 64 + TB - 1) / TB, TB, 0, stream>>>(Wf, packB, HC, 64);
    gatjk_gemm_wmma<<<gemmGridOut, 128, 0, stream>>>(bfA, Nn, HC, packB, (float*)d_out, 64, bf);
}
